// CrossAttention_77489799955185
// MI455X (gfx1250) — matrix operand order repaired and checked by reading, not compiled, not run on hardware
//
#include <hip/hip_runtime.h>

typedef __bf16 bf16;
typedef __attribute__((ext_vector_type(16))) __bf16 v16bf;
typedef __attribute__((ext_vector_type(8)))  float  v8f;

constexpr int BATCH = 8;
constexpr int CH    = 256;
constexpr int HH    = 48;
constexpr int WW    = 48;
constexpr int DD    = 256;
constexpr int NN    = HH * WW;   // 2304

#define DEVFN __device__ __forceinline__

DEVFN v8f wmma_bf16(v16bf a, v16bf b, v8f c) {
  // (neg_a, A, neg_b, B, c_mod, C, reuse_a, reuse_b)
  return __builtin_amdgcn_wmma_f32_16x16x32_bf16(false, a, false, b, (short)0, c,
                                                 false, false);
}

// Async global->LDS 16-byte copy (CDNA5; tracked by ASYNCcnt).
DEVFN void async_b128(unsigned lds_off, const void* gaddr) {
  asm volatile("global_load_async_to_lds_b128 %0, %1, off"
               :: "v"(lds_off), "v"((unsigned long long)gaddr)
               : "memory");
}
DEVFN unsigned lds_off_of(const void* p) {
  // generic shared pointer: addr[31:0] is the LDS byte offset
  return (unsigned)(unsigned long long)p;
}

// ---------------------------------------------------------------------------
// Kernel 1: depthwise 3x3 conv (pad=1) for q/k/v branches; f32 in, bf16 out.
// Output layout [B][N][C] (c fastest) so the 1x1-conv GEMM reads both
// operands K-contiguously with no transposes.
// ---------------------------------------------------------------------------
__global__ __launch_bounds__(256) void dw3_kernel(
    const float* __restrict__ x, const float* __restrict__ y,
    const float* __restrict__ qw, const float* __restrict__ qb,
    const float* __restrict__ kw, const float* __restrict__ kb,
    const float* __restrict__ vw, const float* __restrict__ vb,
    bf16* __restrict__ dq, bf16* __restrict__ dk, bf16* __restrict__ dv) {
  int idx = blockIdx.x * 256 + threadIdx.x;   // covers B*N*C exactly
  int c = idx & (CH - 1);
  int nb = idx >> 8;
  int n = nb % NN;
  int b = nb / NN;
  int h = n / WW, w = n % WW;

  const float* xp = x + ((size_t)b * CH + c) * NN;
  const float* yp = y + ((size_t)b * CH + c) * NN;

  float aq = qb[c], ak = kb[c], av = vb[c];
#pragma unroll
  for (int dy = -1; dy <= 1; ++dy) {
    int hh = h + dy;
    if (hh < 0 || hh >= HH) continue;
#pragma unroll
    for (int dx = -1; dx <= 1; ++dx) {
      int w2 = w + dx;
      if (w2 < 0 || w2 >= WW) continue;
      float xv = xp[hh * WW + w2];
      float yv = yp[hh * WW + w2];
      int wi = c * 9 + (dy + 1) * 3 + (dx + 1);
      aq += qw[wi] * xv;
      av += vw[wi] * xv;
      ak += kw[wi] * yv;
    }
  }
  size_t o = ((size_t)b * NN + n) * CH + c;   // [B][N][C], coalesced over c
  dq[o] = (bf16)aq;
  dk[o] = (bf16)ak;
  dv[o] = (bf16)av;
}

// ---------------------------------------------------------------------------
// Kernel 2: convert 1x1-conv weights [D][C] f32 -> bf16 (all three branches)
// ---------------------------------------------------------------------------
__global__ __launch_bounds__(256) void cvtw_kernel(
    const float* __restrict__ qp, const float* __restrict__ kp,
    const float* __restrict__ vp, bf16* __restrict__ wq,
    bf16* __restrict__ wk, bf16* __restrict__ wv) {
  int i = blockIdx.x * 256 + threadIdx.x;  // D*C = 65536
  wq[i] = (bf16)qp[i];
  wk[i] = (bf16)kp[i];
  wv[i] = (bf16)vp[i];
}

// ---------------------------------------------------------------------------
// Kernel 3: pointwise conv as a pure-register WMMA GEMM (no LDS!).
//   out[n][d] = sum_c X[n][c] * W[d][c] + bias[d], then * scale.
//   A = X tile (16 n x 32 c), rows K-contiguous in memory.
//   B = W    (32 c x 16 d), columns (d rows of W) K-contiguous in memory.
// Block: 256 threads = 8 waves; wave w owns d-tile (blockIdx.y*8 + w).
// v_layout==0 -> store [B][N][D] (Q,K) ; v_layout==1 -> store [B][D][N] (V).
// ---------------------------------------------------------------------------
__global__ __launch_bounds__(256) void pw_kernel(
    const bf16* __restrict__ X,    // [B][N][C]
    const bf16* __restrict__ Wb,   // [D][C] bf16
    const float* __restrict__ bias,
    bf16* __restrict__ out, int v_layout, float scale) {
  const int b = blockIdx.z;
  const int n0 = blockIdx.x * 16;
  const int t = threadIdx.x;
  const int lane = t & 31, wave = t >> 5;
  const int d0 = (blockIdx.y * 8 + wave) * 16;
  const int l15 = lane & 15;
  const int kb8  = (lane >= 16) ? 8 : 0;   // unused (A and B frags both use kh16)
  const int kh16 = (lane >= 16) ? 16 : 0;  // B-frag K-half offset

  const bf16* xrow = X + ((size_t)b * NN + n0 + l15) * CH;
  const bf16* wrow = Wb + (size_t)(d0 + l15) * CH;

  v8f acc = {};
#pragma unroll
  for (int ks = 0; ks < 8; ++ks) {
    const int c0 = ks * 32;
    union { uint4 u[2]; v16bf v; } a, bb;
    a.u[0]  = *(const uint4*)(xrow + c0 + kh16);
    a.u[1]  = *(const uint4*)(xrow + c0 + kh16 + 8);
    bb.u[0] = *(const uint4*)(wrow + c0 + kh16);
    bb.u[1] = *(const uint4*)(wrow + c0 + kh16 + 8);
    acc = wmma_bf16(a.v, bb.v, acc);
  }

  // Epilogue. D-frag: col(d_local)=lane&15, row(n_local)=r + 8*(lane>=16)
  const int rbase = (lane >= 16) ? 8 : 0;
  const int d = d0 + l15;
  const float bias_d = bias[d];
#pragma unroll
  for (int r = 0; r < 8; ++r) {
    int n = n0 + rbase + r;
    float v = (acc[r] + bias_d) * scale;
    bf16 bv = (bf16)v;
    if (v_layout)
      out[((size_t)b * DD + d) * NN + n] = bv;   // V: [B][D][N]
    else
      out[((size_t)b * NN + n) * DD + d] = bv;   // Q,K: [B][N][D] (coalesced)
  }
}

// ---------------------------------------------------------------------------
// Kernel 4: flash attention (QK^T -> online softmax -> PV -> ReLU), with
// double-buffered async global->LDS staging of K/V tiles (ASYNCcnt) and a
// single barrier per 32-key block.
// Block: 256 threads = 8 waves, 128 query rows of one batch (16 per wave).
// ---------------------------------------------------------------------------
__global__ __launch_bounds__(256) void attn_kernel(
    const bf16* __restrict__ Q,   // [B][N][D], pre-scaled by 1/sqrt(D)
    const bf16* __restrict__ Kt,  // [B][N][D]
    const bf16* __restrict__ Vd,  // [B][D][N]
    float* __restrict__ out) {    // [B][D][N]
  __shared__ bf16 ksm[2][32][DD + 8];     // keys x d, double buffered
  __shared__ bf16 vsm[2][DD][32 + 8];     // d x keys, double buffered
  __shared__ bf16 psm[8][16][32 + 8];     // per-wave P staging

  const int b = blockIdx.y;
  const int t = threadIdx.x;
  const int lane = t & 31, wave = t >> 5;
  const int n0 = blockIdx.x * 128 + wave * 16;
  const int rowA = lane & 15;
  const int kb8  = (lane >= 16) ? 8 : 0;
  const int kh16 = (lane >= 16) ? 16 : 0;

  // Q fragments for all 8 d-steps (registers)
  union AF { uint4 u[2]; v16bf v; };
  AF qf[8];
  {
    const bf16* qrow = Q + ((size_t)b * NN + n0 + rowA) * DD;
#pragma unroll
    for (int s = 0; s < 8; ++s) {
      qf[s].u[0] = *(const uint4*)(qrow + s * 32 + kh16);
      qf[s].u[1] = *(const uint4*)(qrow + s * 32 + kh16 + 8);
    }
  }

  v8f O[16];
#pragma unroll
  for (int i = 0; i < 16; ++i) O[i] = (v8f){};
  float Mrow[8], Lrow[8];
#pragma unroll
  for (int r = 0; r < 8; ++r) { Mrow[r] = -1e30f; Lrow[r] = 0.0f; }

  // this thread's cooperative-staging slice
  const int krow = t >> 3;          // 0..31 (key row)
  const int kcol = (t & 7) * 32;    // 0..224 (d col)
  const bf16* kbase = Kt + ((size_t)b * NN + krow) * DD + kcol;
  const bf16* vbase = Vd + ((size_t)b * DD + t) * NN;   // t = d row (0..255)
  constexpr int NBLK = NN / 32;     // 72

  // prologue: stage block 0 into buffer 0
#pragma unroll
  for (int i = 0; i < 4; ++i) {
    async_b128(lds_off_of(&ksm[0][krow][kcol + i * 8]), kbase + i * 8);
    async_b128(lds_off_of(&vsm[0][t][i * 8]),           vbase + i * 8);
  }

  for (int kblk = 0; kblk < NBLK; ++kblk) {
    const int cur = kblk & 1;
    asm volatile("s_wait_asynccnt 0" ::: "memory");
    __syncthreads();  // buffer `cur` fully staged; buffer `1-cur` fully read

    if (kblk + 1 < NBLK) {  // stream next block into the other buffer
      const int m1 = (kblk + 1) * 32;
#pragma unroll
      for (int i = 0; i < 4; ++i) {
        async_b128(lds_off_of(&ksm[1 - cur][krow][kcol + i * 8]),
                   kbase + (size_t)m1 * DD + i * 8);
        async_b128(lds_off_of(&vsm[1 - cur][t][i * 8]), vbase + m1 + i * 8);
      }
    }

    // S tile 16x32 = two 16x16 accumulators, K-reduced over D in 8 steps
    v8f S0 = {}, S1 = {};
#pragma unroll
    for (int s = 0; s < 8; ++s) {
      union { uint4 u[2]; v16bf v; } b0, b1;
      const bf16* k0 = &ksm[cur][rowA][s * 32 + kh16];
      const bf16* k1 = &ksm[cur][16 + rowA][s * 32 + kh16];
      b0.u[0] = *(const uint4*)k0;  b0.u[1] = *(const uint4*)(k0 + 8);
      b1.u[0] = *(const uint4*)k1;  b1.u[1] = *(const uint4*)(k1 + 8);
      S0 = wmma_bf16(qf[s].v, b0.v, S0);
      S1 = wmma_bf16(qf[s].v, b1.v, S1);
    }

    // Online softmax (rows live in 16-lane halves -> shfl_xor 1,2,4,8)
#pragma unroll
    for (int r = 0; r < 8; ++r) {
      float m = fmaxf(S0[r], S1[r]);
      m = fmaxf(m, __shfl_xor(m, 1, 32));
      m = fmaxf(m, __shfl_xor(m, 2, 32));
      m = fmaxf(m, __shfl_xor(m, 4, 32));
      m = fmaxf(m, __shfl_xor(m, 8, 32));
      float Mn = fmaxf(Mrow[r], m);
      float alpha = __expf(Mrow[r] - Mn);
      Mrow[r] = Mn;
      Lrow[r] *= alpha;
#pragma unroll
      for (int dt = 0; dt < 16; ++dt) O[dt][r] *= alpha;
      float p0 = __expf(S0[r] - Mn);
      float p1 = __expf(S1[r] - Mn);
      S0[r] = p0;
      S1[r] = p1;
      float rs = p0 + p1;
      rs += __shfl_xor(rs, 1, 32);
      rs += __shfl_xor(rs, 2, 32);
      rs += __shfl_xor(rs, 4, 32);
      rs += __shfl_xor(rs, 8, 32);
      Lrow[r] += rs;
    }

    // P: C/D layout -> per-wave LDS -> A-fragment layout (bf16)
    {
      int colp = lane & 15;
#pragma unroll
      for (int r = 0; r < 8; ++r) {
        int rw = r + kb8;
        psm[wave][rw][colp]      = (bf16)S0[r];
        psm[wave][rw][colp + 16] = (bf16)S1[r];
      }
    }
    asm volatile("s_wait_dscnt 0" ::: "memory");  // same-wave LDS RAW fence
    union { uint4 u[2]; v16bf v; } pa;
    {
      const bf16* pr = &psm[wave][rowA][0];
      pa.u[0] = *(const uint4*)(pr + kh16);
      pa.u[1] = *(const uint4*)(pr + kh16 + 8);
    }

    // O += P(16x32) x V(32x16) for each of the 16 d-column tiles
#pragma unroll
    for (int dt = 0; dt < 16; ++dt) {
      union { uint4 u[2]; v16bf v; } vb;
      const bf16* vr = &vsm[cur][dt * 16 + rowA][kh16];
      vb.u[0] = *(const uint4*)vr;
      vb.u[1] = *(const uint4*)(vr + 8);
      O[dt] = wmma_bf16(pa.v, vb.v, O[dt]);
    }
    // no end-of-iteration barrier needed: next iteration's top barrier
    // orders reads of buf[cur] against the async writes into it.
  }

  // Epilogue: normalize by L, ReLU, store f32 to [B][D][N]
  float inv[8];
#pragma unroll
  for (int r = 0; r < 8; ++r) inv[r] = 1.0f / Lrow[r];
#pragma unroll
  for (int dt = 0; dt < 16; ++dt) {
#pragma unroll
    for (int r = 0; r < 8; ++r) {
      float v = fmaxf(O[dt][r] * inv[r], 0.0f);
      int d = dt * 16 + rowA;
      int n = n0 + r + kb8;
      out[((size_t)b * DD + d) * NN + n] = v;
    }
  }
}

// ---------------------------------------------------------------------------
extern "C" void kernel_launch(void* const* d_in, const int* in_sizes, int n_in,
                              void* d_out, int out_size, void* d_ws,
                              size_t ws_size, hipStream_t stream) {
  const float* x    = (const float*)d_in[0];
  const float* y    = (const float*)d_in[1];
  const float* qd_w = (const float*)d_in[2];
  const float* qd_b = (const float*)d_in[3];
  const float* qp_w = (const float*)d_in[4];
  const float* qp_b = (const float*)d_in[5];
  const float* kd_w = (const float*)d_in[6];
  const float* kd_b = (const float*)d_in[7];
  const float* kp_w = (const float*)d_in[8];
  const float* kp_b = (const float*)d_in[9];
  const float* vd_w = (const float*)d_in[10];
  const float* vd_b = (const float*)d_in[11];
  const float* vp_w = (const float*)d_in[12];
  const float* vp_b = (const float*)d_in[13];
  float* out = (float*)d_out;

  char* ws = (char*)d_ws;
  const size_t S1 = (size_t)BATCH * CH * NN * sizeof(bf16);  // 9.44 MB
  bf16* dq = (bf16*)(ws + 0 * S1);  // [B][N][C]
  bf16* dk = (bf16*)(ws + 1 * S1);
  bf16* dv = (bf16*)(ws + 2 * S1);
  bf16* Qm = (bf16*)(ws + 3 * S1);  // [B][N][D], pre-scaled by 1/sqrt(D)
  bf16* Km = (bf16*)(ws + 4 * S1);  // [B][N][D]
  bf16* Vm = (bf16*)(ws + 5 * S1);  // [B][D][N]
  bf16* wq = (bf16*)(ws + 6 * S1);
  bf16* wk = wq + (size_t)DD * CH;
  bf16* wv = wk + (size_t)DD * CH;

  dw3_kernel<<<BATCH * CH * NN / 256, 256, 0, stream>>>(
      x, y, qd_w, qd_b, kd_w, kd_b, vd_w, vd_b, dq, dk, dv);
  cvtw_kernel<<<DD * CH / 256, 256, 0, stream>>>(qp_w, kp_w, vp_w, wq, wk, wv);

  dim3 g2(NN / 16, DD / 128, BATCH);  // 144 x 2 x 8 blocks, 8 waves each
  pw_kernel<<<g2, 256, 0, stream>>>(dq, wq, qp_b, Qm, 0, 0.0625f);  // 1/sqrt(256)
  pw_kernel<<<g2, 256, 0, stream>>>(dk, wk, kp_b, Km, 0, 1.0f);
  pw_kernel<<<g2, 256, 0, stream>>>(dv, wv, vp_b, Vm, 1, 1.0f);

  dim3 g3(NN / 128, BATCH);
  attn_kernel<<<g3, 256, 0, stream>>>(Qm, Km, Vm, out);
}
